// OnlineFlashMTPModel_45122926412364
// MI455X (gfx1250) — compile-verified, matrix operand order checked by reading, not run on hardware
//
#include <hip/hip_runtime.h>
#include <hip/hip_bf16.h>
#include <stdint.h>

// ---------------------------------------------------------------------------
// MI455X (gfx1250) implementation. wave32, WMMA bf16 16x16x32 everywhere,
// async-to-LDS double-buffered GEMM staging (ASYNCcnt), NT stores for logits.
constexpr int S_  = 2048;
constexpr int N_  = 64;
constexpr int BS_ = 16;
constexpr int W_  = 512;
constexpr int NH_ = 8;
constexpr int H_  = 1024;
constexpr int V_  = 32000;
constexpr int L_  = 2;
constexpr int DH_ = H_ / NH_;          // 128
constexpr int T_  = S_ + N_ * BS_;     // 3072

typedef __bf16 bf16_t;
typedef __attribute__((ext_vector_type(16))) __bf16 v16bf;
typedef __attribute__((ext_vector_type(8)))  __bf16 v8bf;
typedef __attribute__((ext_vector_type(8)))  float  v8f;

// Async copy 16B global -> LDS (gfx1250 GLOBAL_LOAD_ASYNC_TO_LDS_B128, ASYNCcnt)
__device__ __forceinline__ void async_ld_b128(uint32_t lds_off, const void* gptr)
{
    asm volatile("global_load_async_to_lds_b128 %0, %1, off"
                 :: "v"(lds_off), "v"(gptr)
                 : "memory");
}
__device__ __forceinline__ void wait_async_3()
{
    asm volatile("s_wait_asynccnt 0x3" ::: "memory");
}
__device__ __forceinline__ void wait_async_0()
{
    asm volatile("s_wait_asynccnt 0x0" ::: "memory");
}

// ---------------------------------------------------------------------------
// fp32 -> bf16 transpose-convert: src[K x N] row-major -> dst[N x K] row-major.
__global__ void __launch_bounds__(256) k_transpose_bf16(
    const float* __restrict__ src, bf16_t* __restrict__ dst, int K, int N)
{
    size_t idx = (size_t)blockIdx.x * blockDim.x + threadIdx.x;
    size_t total = (size_t)K * N;
    if (idx >= total) return;
    int k = (int)(idx / N);
    int n = (int)(idx % N);
    dst[(size_t)n * K + k] = (bf16_t)src[idx];
}

__global__ void __launch_bounds__(256) k_f32_to_bf16(
    const float* __restrict__ src, bf16_t* __restrict__ dst, size_t total)
{
    size_t idx = (size_t)blockIdx.x * blockDim.x + threadIdx.x;
    if (idx < total) dst[idx] = (bf16_t)src[idx];
}

// ---------------------------------------------------------------------------
// target_hidden gather: th[n][c] = hidden[layer(c)][ctx(n)][c%H]  (bf16, 64x2048)
__global__ void __launch_bounds__(256) k_build_th(
    const float* __restrict__ hidden, const int* __restrict__ anchors,
    bf16_t* __restrict__ th)
{
    int idx = blockIdx.x * blockDim.x + threadIdx.x;
    if (idx >= N_ * L_ * H_) return;
    int n = idx >> 11;
    int c = idx & (L_ * H_ - 1);
    int l = c >> 10;
    int hc = c & (H_ - 1);
    int ctx = anchors[n] - 1; if (ctx < 0) ctx = 0;
    th[idx] = (bf16_t)hidden[((size_t)l * S_ + ctx) * H_ + hc];
}

// ---------------------------------------------------------------------------
// h = concat(embed[input_ids], embed[block_ids] + inj_broadcast)   (fp32 T x H)
__global__ void __launch_bounds__(256) k_build_h(
    const int* __restrict__ ids, const int* __restrict__ anchors,
    const unsigned char* __restrict__ keep,
    const float* __restrict__ embed, const float* __restrict__ inj,
    float* __restrict__ h)
{
    size_t idx = (size_t)blockIdx.x * blockDim.x + threadIdx.x;
    if (idx >= (size_t)T_ * H_) return;
    int t = (int)(idx >> 10);
    int c = (int)(idx & (H_ - 1));
    float v;
    if (t < S_) {
        v = embed[(size_t)ids[t] * H_ + c];
    } else {
        int r = t - S_;
        int b = r >> 4;
        int off = r & 15;
        int tok = 0;   // MASK_TOKEN
        if (off == 0 && keep[b]) {
            int ap = anchors[b];
            if (ap < 0) ap = 0; if (ap > S_ - 1) ap = S_ - 1;
            tok = ids[ap];
        }
        v = embed[(size_t)tok * H_ + c] + inj[(size_t)b * H_ + c];
    }
    h[idx] = v;
}

// ---------------------------------------------------------------------------
// rmsnorm row kernel: xb[t][c] = bf16( h[t][c] * rsqrt(mean(h^2)+eps) * w[c] )
__global__ void __launch_bounds__(256) k_rmsnorm_bf16(
    const float* __restrict__ h, const float* __restrict__ w,
    bf16_t* __restrict__ xb)
{
    __shared__ float red[8];
    int t = blockIdx.x;
    int tid = threadIdx.x;
    const float* row = h + (size_t)t * H_;
    float ss = 0.f;
    for (int c = tid; c < H_; c += 256) { float v = row[c]; ss += v * v; }
    #pragma unroll
    for (int off = 16; off > 0; off >>= 1) ss += __shfl_xor(ss, off, 32);
    if ((tid & 31) == 0) red[tid >> 5] = ss;
    __syncthreads();
    if (tid < 32) {
        float v = (tid < 8) ? red[tid] : 0.f;
        v += __shfl_xor(v, 4, 32);
        v += __shfl_xor(v, 2, 32);
        v += __shfl_xor(v, 1, 32);
        if (tid == 0) red[0] = v;
    }
    __syncthreads();
    float rs = rsqrtf(red[0] / (float)H_ + 1e-6f);
    for (int c = tid; c < H_; c += 256)
        xb[(size_t)t * H_ + c] = (bf16_t)(row[c] * rs * w[c]);
}

// ---------------------------------------------------------------------------
// RoPE on q/k + convert to bf16.
__global__ void __launch_bounds__(256) k_rope_bf16(
    const float* __restrict__ qf, const float* __restrict__ kf,
    const int* __restrict__ anchors,
    bf16_t* __restrict__ qb, bf16_t* __restrict__ kb)
{
    int idx = blockIdx.x * blockDim.x + threadIdx.x;
    if (idx >= T_ * NH_ * (DH_ / 2)) return;
    int i = idx & 63;
    int tmp = idx >> 6;
    int head = tmp & (NH_ - 1);
    int t = tmp >> 3;
    int pos = (t < S_) ? t : (anchors[(t - S_) >> 4] + ((t - S_) & 15));
    float inv = __powf(10000.f, -(float)(2 * i) / (float)DH_);
    float fr = (float)pos * inv;
    float c = __cosf(fr), s = __sinf(fr);
    size_t base = (size_t)t * H_ + head * DH_;
    float x1 = qf[base + i], x2 = qf[base + 64 + i];
    qb[base + i]      = (bf16_t)(x1 * c - x2 * s);
    qb[base + 64 + i] = (bf16_t)(x2 * c + x1 * s);
    x1 = kf[base + i]; x2 = kf[base + 64 + i];
    kb[base + i]      = (bf16_t)(x1 * c - x2 * s);
    kb[base + 64 + i] = (bf16_t)(x2 * c + x1 * s);
}

// ---------------------------------------------------------------------------
// WMMA GEMM with async-to-LDS double-buffered staging.
//   C[M x Nout] = A[M x K](bf16) * BT[Nout x K](bf16)^T (+addend)
// Block: 256 threads = 8 waves; block tile 64(M) x 128(N); wave tile 16 x 64.
// Per K-step(32): stage A(64x32, 4KB) + B(128x32, 8KB) via
// global_load_async_to_lds_b128 (3 async instrs/wave), double buffered.
// Steady-state loop is branch-free: stage(next) -> s_wait_asynccnt 0x3 ->
// barrier -> [10x ds_load_b128, one wait, 4x wmma] -> barrier.
// Requires: M % 64 == 0, Nout % 128 == 0, K % 32 == 0.
__global__ void __launch_bounds__(256) k_gemm_bf16(
    const bf16_t* __restrict__ A, const bf16_t* __restrict__ BT,
    float* __restrict__ C, const float* __restrict__ addend,
    int M, int K, int Nout)
{
    __shared__ bf16_t ldsA[2][64 * 32];    // [buf][row*32 + k]
    __shared__ bf16_t ldsB[2][128 * 32];

    const int tid  = threadIdx.x;
    const int lane = tid & 31;
    const int wave = tid >> 5;
    const int hf   = lane >> 4;
    const int l15  = lane & 15;
    const int mwave = wave & 3;            // 4 sub-tiles along M
    const int nwave = wave >> 2;           // 2 sub-tiles along N
    const int mblock = blockIdx.y * 64;
    const int nblock = blockIdx.x * 128;

    // Per-thread staging assignments (16B chunks along K)
    const int arow = tid >> 2;             // 0..63
    const int akp  = (tid & 3) * 8;        // bf16 elems
    const int brow0 = (tid * 2) >> 2;      // two B chunks per thread
    const int bkp0  = ((tid * 2) & 3) * 8;
    const int brow1 = (tid * 2 + 1) >> 2;
    const int bkp1  = ((tid * 2 + 1) & 3) * 8;

    auto stage = [&](int buf, int kk) {
        uint32_t aoff = (uint32_t)(uintptr_t)&ldsA[buf][arow * 32 + akp];
        async_ld_b128(aoff, (const void*)(A + (size_t)(mblock + arow) * K + kk + akp));
        uint32_t b0 = (uint32_t)(uintptr_t)&ldsB[buf][brow0 * 32 + bkp0];
        async_ld_b128(b0, (const void*)(BT + (size_t)(nblock + brow0) * K + kk + bkp0));
        uint32_t b1 = (uint32_t)(uintptr_t)&ldsB[buf][brow1 * 32 + bkp1];
        async_ld_b128(b1, (const void*)(BT + (size_t)(nblock + brow1) * K + kk + bkp1));
        // warm L2 for the K-step after next (prefetch ahead of the async engine)
        __builtin_prefetch((const void*)(BT + (size_t)(nblock + brow0) * K + kk + 64), 0, 1);
    };

    v8f acc[4];
    #pragma unroll
    for (int s = 0; s < 4; ++s) acc[s] = (v8f){0,0,0,0,0,0,0,0};

    auto compute = [&](int cur) {
        // Load ALL fragments first (one DS wait), then WMMA back-to-back.
        v16bf a, b[4];
        {
            const bf16_t* ar = &ldsA[cur][(mwave * 16 + l15) * 32];
            v8bf lo = *(const v8bf*)(ar + 8 * hf);
            v8bf hi = *(const v8bf*)(ar + 16 + 8 * hf);
            #pragma unroll
            for (int j = 0; j < 8; ++j) { a[j] = lo[j]; a[8 + j] = hi[j]; }
        }
        #pragma unroll
        for (int s = 0; s < 4; ++s) {
            const bf16_t* br = &ldsB[cur][(nwave * 64 + s * 16 + l15) * 32];
            v8bf lo = *(const v8bf*)(br + 8 * hf);
            v8bf hi = *(const v8bf*)(br + 16 + 8 * hf);
            #pragma unroll
            for (int j = 0; j < 8; ++j) { b[s][j] = lo[j]; b[s][8 + j] = hi[j]; }
        }
        #pragma unroll
        for (int s = 0; s < 4; ++s)
            acc[s] = __builtin_amdgcn_wmma_f32_16x16x32_bf16(
                false, a, false, b[s], (short)0, acc[s], false, false);
    };

    stage(0, 0);
    int kk = 0;
    const int last = K - 32;
    for (; kk < last; kk += 32) {                 // branch-free steady state
        const int cur = (kk >> 5) & 1;
        stage(cur ^ 1, kk + 32);
        wait_async_3();                           // previous stage complete
        __syncthreads();
        compute(cur);
        __syncthreads();                          // readers done before re-stage
    }
    wait_async_0();                               // peeled last K-step
    __syncthreads();
    compute((last >> 5) & 1);

    // C/D layout: lane l holds (M = vr + 8*(l>>4), N = l&15) in acc element vr.
    #pragma unroll
    for (int s = 0; s < 4; ++s)
        #pragma unroll
        for (int vr = 0; vr < 8; ++vr) {
            int row = mblock + mwave * 16 + vr + 8 * hf;
            int col = nblock + nwave * 64 + s * 16 + l15;
            size_t idx = (size_t)row * Nout + col;
            float v = acc[s][vr];
            if (addend) v += addend[idx];
            __builtin_nontemporal_store(v, &C[idx]);   // NT: keep L2 for weights
        }
}

// ---------------------------------------------------------------------------
// Flash attention, one wave per (16-row q tile, head). Online softmax.
// V-chunk staged with async-to-LDS so the DMA overlaps QK^T WMMAs + softmax;
// a single s_wait_asynccnt 0x0 gates the P*V phase.
__global__ void __launch_bounds__(32) k_flash_attn(
    const bf16_t* __restrict__ qb, const bf16_t* __restrict__ kb,
    const bf16_t* __restrict__ vb,
    const int* __restrict__ anchors, const unsigned char* __restrict__ keep,
    bf16_t* __restrict__ ob)
{
    __shared__ bf16_t lds_p[16 * 32];     // prob tile C-layout -> A-layout relay
    __shared__ bf16_t lds_v[32 * DH_];    // staged V rows for the chunk

    const int qt = blockIdx.x, head = blockIdx.y;
    const int lane = threadIdx.x;
    const int hf = lane >> 4, l15 = lane & 15;
    const int qbase = qt * 16;
    const float scale = 0.08838834764831845f;   // 1/sqrt(128)

    v16bf qa[4];
    {
        const bf16_t* qrow = qb + (size_t)(qbase + l15) * H_ + head * DH_;
        #pragma unroll
        for (int t = 0; t < 4; ++t) {
            v8bf lo = *(const v8bf*)(qrow + t * 32 + 8 * hf);
            v8bf hi = *(const v8bf*)(qrow + t * 32 + 16 + 8 * hf);
            #pragma unroll
            for (int j = 0; j < 8; ++j) { qa[t][j] = lo[j]; qa[t][8 + j] = hi[j]; }
        }
    }

    bool inFullQ[8]; int anchorQ[8], blkQ[8], qg_[8]; bool validQ[8];
    #pragma unroll
    for (int vr = 0; vr < 8; ++vr) {
        int qg = qbase + vr + 8 * hf; qg_[vr] = qg;
        bool inf = qg < S_; inFullQ[vr] = inf;
        int b = inf ? 0 : ((qg - S_) >> 4); if (b > N_ - 1) b = N_ - 1;
        blkQ[vr] = b;
        anchorQ[vr] = anchors[b];
        validQ[vr] = inf || (keep[b] != 0);
    }

    float mrow[8], lrow[8];
    v8f oaccv[8];
    #pragma unroll
    for (int vr = 0; vr < 8; ++vr) { mrow[vr] = -3.0e38f; lrow[vr] = 0.f; }
    #pragma unroll
    for (int s = 0; s < 8; ++s) oaccv[s] = (v8f){0,0,0,0,0,0,0,0};

    auto process_chunk = [&](int kbase) {
        // Kick off async DMA of 32 V rows (overlaps the QK^T WMMAs below)
        {
            int key = kbase + lane; if (key > T_ - 1) key = T_ - 1;
            const bf16_t* vrow = vb + (size_t)key * H_ + head * DH_;
            uint32_t loff = (uint32_t)(uintptr_t)&lds_v[lane * DH_];
            #pragma unroll
            for (int j = 0; j < 16; ++j)
                async_ld_b128(loff + j * 16, (const void*)(vrow + j * 8));
        }
        // scores S = Q K^T  (two 16x16 key tiles, K-dim = 128 -> 4 WMMAs each)
        v8f sc[2];
        #pragma unroll
        for (int t = 0; t < 2; ++t) {
            v8f acc = (v8f){0,0,0,0,0,0,0,0};
            int key = kbase + t * 16 + l15;
            int keyc = key > T_ - 1 ? T_ - 1 : key;
            const bf16_t* krow = kb + (size_t)keyc * H_ + head * DH_;
            #pragma unroll
            for (int kk = 0; kk < 4; ++kk) {
                v16bf bt;
                v8bf lo = *(const v8bf*)(krow + kk * 32 + 8 * hf);
                v8bf hi = *(const v8bf*)(krow + kk * 32 + 16 + 8 * hf);
                #pragma unroll
                for (int j = 0; j < 8; ++j) { bt[j] = lo[j]; bt[8 + j] = hi[j]; }
                acc = __builtin_amdgcn_wmma_f32_16x16x32_bf16(
                    false, qa[kk], false, bt, (short)0, acc, false, false);
            }
            sc[t] = acc;
        }
        // mask
        #pragma unroll
        for (int t = 0; t < 2; ++t) {
            int kg = kbase + t * 16 + l15;
            #pragma unroll
            for (int vr = 0; vr < 8; ++vr) {
                float s = sc[t][vr] * scale;
                bool ok;
                if (!validQ[vr] || kg >= T_) ok = false;
                else if (inFullQ[vr]) ok = (kg < S_) && (kg <= qg_[vr]);
                else {
                    int a = anchorQ[vr];
                    int wstart = a - W_ + 1; if (wstart < 0) wstart = 0;
                    bool see_full = (kg < S_) && (kg >= wstart) && (kg <= a - 1);
                    bool bidir = (kg >= S_) && (((kg - S_) >> 4) == blkQ[vr]);
                    ok = see_full || bidir;
                }
                sc[t][vr] = ok ? s : -1.0e9f;
            }
        }
        // online softmax update per row, probs -> LDS (bf16)
        #pragma unroll
        for (int vr = 0; vr < 8; ++vr) {
            float mt = fmaxf(sc[0][vr], sc[1][vr]);
            #pragma unroll
            for (int off = 1; off < 16; off <<= 1) mt = fmaxf(mt, __shfl_xor(mt, off, 32));
            float mn = fmaxf(mrow[vr], mt);
            float corr = __expf(mrow[vr] - mn);
            float p0 = __expf(sc[0][vr] - mn);
            float p1 = __expf(sc[1][vr] - mn);
            float ps = p0 + p1;
            #pragma unroll
            for (int off = 1; off < 16; off <<= 1) ps += __shfl_xor(ps, off, 32);
            mrow[vr] = mn;
            lrow[vr] = lrow[vr] * corr + ps;
            #pragma unroll
            for (int s = 0; s < 8; ++s) oaccv[s][vr] *= corr;
            lds_p[(vr + 8 * hf) * 32 + l15]      = (bf16_t)p0;
            lds_p[(vr + 8 * hf) * 32 + 16 + l15] = (bf16_t)p1;
        }
        wait_async_0();          // V DMA complete (overlapped with all of the above)
        __syncthreads();
        // reload probs in A-layout (16 x 32)
        v16bf pa;
        {
            const bf16_t* prow = &lds_p[l15 * 32];
            v8bf lo = *(const v8bf*)(prow + 8 * hf);
            v8bf hi = *(const v8bf*)(prow + 16 + 8 * hf);
            #pragma unroll
            for (int j = 0; j < 8; ++j) { pa[j] = lo[j]; pa[8 + j] = hi[j]; }
        }
        // O += P V : 8 dim-slices of 16, B-tile gathered from LDS
        #pragma unroll
        for (int s = 0; s < 8; ++s) {
            v16bf bv;
            #pragma unroll
            for (int j = 0; j < 8; ++j) {
                bv[j]     = lds_v[(8 * hf + j) * DH_ + s * 16 + l15];
                bv[8 + j] = lds_v[(16 + 8 * hf + j) * DH_ + s * 16 + l15];
            }
            oaccv[s] = __builtin_amdgcn_wmma_f32_16x16x32_bf16(
                false, pa, false, bv, (short)0, oaccv[s], false, false);
        }
        __syncthreads();
    };

    if (qbase + 16 <= S_) {
        int kmax = qbase + 15;                        // causal full-region tile
        for (int kb = 0; kb <= kmax; kb += 32) process_chunk(kb);
    } else {
        int b = qt - S_ / 16;                          // block-region tile
        int a = anchors[b];
        int ws = a - W_ + 1; if (ws < 0) ws = 0;
        int we = a - 1;
        for (int kb = ws & ~31; kb <= we; kb += 32) process_chunk(kb);
        process_chunk(S_ + b * 16);                    // own bidirectional block
    }

    #pragma unroll
    for (int s = 0; s < 8; ++s)
        #pragma unroll
        for (int vr = 0; vr < 8; ++vr) {
            float l = lrow[vr]; if (!(l > 0.f)) l = 1.f;
            ob[(size_t)(qbase + vr + 8 * hf) * H_ + head * DH_ + s * 16 + l15] =
                (bf16_t)(oaccv[s][vr] / l);
        }
}

// ---------------------------------------------------------------------------
extern "C" void kernel_launch(void* const* d_in, const int* in_sizes, int n_in,
                              void* d_out, int out_size, void* d_ws, size_t ws_size,
                              hipStream_t stream)
{
    const int*   input_ids = (const int*)  d_in[0];
    const float* hidden    = (const float*)d_in[1];
    const int*   anchors   = (const int*)  d_in[2];
    const unsigned char* keep = (const unsigned char*)d_in[3];   // bool array
    const float* Wembed = (const float*)d_in[4];
    const float* Wq = (const float*)d_in[5];
    const float* Wk = (const float*)d_in[6];
    const float* Wv = (const float*)d_in[7];
    const float* Wo = (const float*)d_in[8];
    const float* fc = (const float*)d_in[9];
    const float* lm = (const float*)d_in[10];
    const float* nw = (const float*)d_in[11];
    float* out = (float*)d_out;

    uint8_t* p = (uint8_t*)d_ws;
    auto carve = [&](size_t bytes) -> void* {
        void* r = (void*)p;
        p += (bytes + 255) & ~(size_t)255;
        return r;
    };
    bf16_t* WqT = (bf16_t*)carve((size_t)H_ * H_ * 2);
    bf16_t* WkT = (bf16_t*)carve((size_t)H_ * H_ * 2);
    bf16_t* WvT = (bf16_t*)carve((size_t)H_ * H_ * 2);
    bf16_t* WoT = (bf16_t*)carve((size_t)H_ * H_ * 2);
    bf16_t* fcT = (bf16_t*)carve((size_t)H_ * (L_ * H_) * 2);      // [1024 x 2048]
    bf16_t* lmT = (bf16_t*)carve((size_t)V_ * H_ * 2);             // [32000 x 1024]
    bf16_t* th  = (bf16_t*)carve((size_t)N_ * L_ * H_ * 2);        // [64 x 2048]
    float*  inj = (float*) carve((size_t)N_ * H_ * 4);             // [64 x 1024]
    float*  h   = (float*) carve((size_t)T_ * H_ * 4);
    bf16_t* xb  = (bf16_t*)carve((size_t)T_ * H_ * 2);
    float*  qf  = (float*) carve((size_t)T_ * H_ * 4);
    float*  kf  = (float*) carve((size_t)T_ * H_ * 4);
    float*  vf  = (float*) carve((size_t)T_ * H_ * 4);
    bf16_t* qbb = (bf16_t*)carve((size_t)T_ * H_ * 2);
    bf16_t* kbb = (bf16_t*)carve((size_t)T_ * H_ * 2);
    bf16_t* vbb = (bf16_t*)carve((size_t)T_ * H_ * 2);
    bf16_t* obb = (bf16_t*)carve((size_t)T_ * H_ * 2);
    bf16_t* x2b = (bf16_t*)carve((size_t)T_ * H_ * 2);

    auto blocks = [](size_t n) { return (unsigned)((n + 255) / 256); };

    // 1) weight transpose+convert to bf16 [Nout x K]
    hipLaunchKernelGGL(k_transpose_bf16, dim3(blocks((size_t)H_ * H_)), dim3(256), 0, stream, Wq, WqT, H_, H_);
    hipLaunchKernelGGL(k_transpose_bf16, dim3(blocks((size_t)H_ * H_)), dim3(256), 0, stream, Wk, WkT, H_, H_);
    hipLaunchKernelGGL(k_transpose_bf16, dim3(blocks((size_t)H_ * H_)), dim3(256), 0, stream, Wv, WvT, H_, H_);
    hipLaunchKernelGGL(k_transpose_bf16, dim3(blocks((size_t)H_ * H_)), dim3(256), 0, stream, Wo, WoT, H_, H_);
    hipLaunchKernelGGL(k_transpose_bf16, dim3(blocks((size_t)(L_ * H_) * H_)), dim3(256), 0, stream, fc, fcT, L_ * H_, H_);
    hipLaunchKernelGGL(k_transpose_bf16, dim3(blocks((size_t)H_ * V_)), dim3(256), 0, stream, lm, lmT, H_, V_);

    // 2) target_hidden gather, 3) inj = th @ fc_w  (64 x 2048 x 1024)
    hipLaunchKernelGGL(k_build_th, dim3(blocks((size_t)N_ * L_ * H_)), dim3(256), 0, stream, hidden, anchors, th);
    hipLaunchKernelGGL(k_gemm_bf16, dim3(H_ / 128, N_ / 64), dim3(256), 0, stream,
                       th, fcT, inj, (const float*)nullptr, N_, L_ * H_, H_);

    // 4) h = embeddings (+inj on block region)
    hipLaunchKernelGGL(k_build_h, dim3(blocks((size_t)T_ * H_)), dim3(256), 0, stream,
                       input_ids, anchors, keep, Wembed, inj, h);

    // 5) x = rmsnorm(h) -> bf16
    hipLaunchKernelGGL(k_rmsnorm_bf16, dim3(T_), dim3(256), 0, stream, h, nw, xb);

    // 6) QKV GEMMs (3072 x 1024 x 1024 each)
    hipLaunchKernelGGL(k_gemm_bf16, dim3(H_ / 128, T_ / 64), dim3(256), 0, stream,
                       xb, WqT, qf, (const float*)nullptr, T_, H_, H_);
    hipLaunchKernelGGL(k_gemm_bf16, dim3(H_ / 128, T_ / 64), dim3(256), 0, stream,
                       xb, WkT, kf, (const float*)nullptr, T_, H_, H_);
    hipLaunchKernelGGL(k_gemm_bf16, dim3(H_ / 128, T_ / 64), dim3(256), 0, stream,
                       xb, WvT, vf, (const float*)nullptr, T_, H_, H_);

    // 7) RoPE(q,k) -> bf16 ; v -> bf16
    hipLaunchKernelGGL(k_rope_bf16, dim3(blocks((size_t)T_ * NH_ * (DH_ / 2))), dim3(256), 0, stream,
                       qf, kf, anchors, qbb, kbb);
    hipLaunchKernelGGL(k_f32_to_bf16, dim3(blocks((size_t)T_ * H_)), dim3(256), 0, stream,
                       vf, vbb, (size_t)T_ * H_);

    // 8) flash attention: one wave per (q-tile, head)
    hipLaunchKernelGGL(k_flash_attn, dim3(T_ / 16, NH_), dim3(32), 0, stream,
                       qbb, kbb, vbb, anchors, keep, obb);

    // 9) h = h + attn_out @ Wo  (fused residual add in GEMM epilogue)
    hipLaunchKernelGGL(k_gemm_bf16, dim3(H_ / 128, T_ / 64), dim3(256), 0, stream,
                       obb, WoT, h, h, T_, H_, H_);

    // 10) x2 = rmsnorm(h) -> bf16
    hipLaunchKernelGGL(k_rmsnorm_bf16, dim3(T_), dim3(256), 0, stream, h, nw, x2b);

    // 11) logits = x2 @ lm_head  (3072 x 1024 x 32000) -> d_out fp32
    hipLaunchKernelGGL(k_gemm_bf16, dim3(V_ / 128, T_ / 64), dim3(256), 0, stream,
                       x2b, lmT, out, (const float*)nullptr, T_, H_, V_);
}
